// net_new_37228776522058
// MI455X (gfx1250) — matrix operand order repaired and checked by reading, not compiled, not run on hardware
//
#include <hip/hip_runtime.h>
#include <hip/hip_bf16.h>

typedef __attribute__((ext_vector_type(16))) _Float16 v16h;
typedef __attribute__((ext_vector_type(8)))  float    v8f;

#define BATCH 32
#define IMG_H 256
#define IMG_W 256

// ---------------------------------------------------------------------------
// Direct 3x3 conv (pad=1) + folded-BN bias + ReLU. Handles channel-concat
// inputs via two source pointers (CA channels from ina, CB channels from inb).
// ---------------------------------------------------------------------------
template<int CA, int CB, int COUT, int STRIDE>
__global__ void conv3x3_bn_relu(const float* __restrict__ ina,
                                const float* __restrict__ inb,
                                const float* __restrict__ w,
                                const float* __restrict__ bias,
                                float* __restrict__ out,
                                int H, int W)
{
    const int Ho = H / STRIDE, Wo = W / STRIDE;
    const int CIN = CA + CB;
    int idx = blockIdx.x * blockDim.x + threadIdx.x;
    const int total = BATCH * COUT * Ho * Wo;
    if (idx >= total) return;

    int x  = idx % Wo;
    int t  = idx / Wo;
    int y  = t % Ho;  t /= Ho;
    int co = t % COUT;
    int b  = t / COUT;

    float acc = bias[co];
    const float* wp = w + (size_t)co * CIN * 9;

    for (int ci = 0; ci < CIN; ++ci) {
        const float* src = (ci < CA)
            ? (ina + ((size_t)(b * CA + ci)) * H * W)
            : (inb + ((size_t)(b * CB + (ci - CA))) * H * W);
        const float* wk = wp + ci * 9;
        #pragma unroll
        for (int ky = 0; ky < 3; ++ky) {
            int iy = y * STRIDE + ky - 1;
            if (iy < 0 || iy >= H) continue;
            #pragma unroll
            for (int kx = 0; kx < 3; ++kx) {
                int ix = x * STRIDE + kx - 1;
                if (ix < 0 || ix >= W) continue;
                acc = fmaf(src[iy * W + ix], wk[ky * 3 + kx], acc);
            }
        }
    }
    out[idx] = fmaxf(acc, 0.0f);
}

// ---------------------------------------------------------------------------
// Async global->LDS staging of one 32-row x 32-k f32 slab of X.
// 512 b64 chunks (8 B each), 128 threads -> 4 chunks/thread.
// LDS row stride = 34 floats (136 B): 8 B aligned for b64 async writes, and
// 34*m mod 64 is distinct/even for m=0..15 -> bank-conflict-free ds_load_b64
// when 16 lanes later read 16 different rows at the same k offset.
// Tracked with ASYNCcnt (gfx1250 async-copy path).
// ---------------------------------------------------------------------------
#define LDS_ROW 34

template<int K>
__device__ inline void async_stage_slab(const float* __restrict__ X, int kk,
                                        unsigned int lds_base, int t)
{
    #pragma unroll
    for (int i = 0; i < 4; ++i) {
        int c   = t + i * 128;        // 0..511
        int row = c >> 4;             // 0..31
        int seg = c & 15;             // 0..15 (8-byte segments in a 32f row)
        const float* gp = X + (size_t)row * K + kk + seg * 2;
        unsigned int lo = lds_base + (unsigned)(row * (LDS_ROW * 4) + seg * 8);
        asm volatile("global_load_async_to_lds_b64 %0, %1, off"
                     :: "v"(lo), "v"(gp) : "memory");
    }
}

__device__ inline void wait_async_all()
{
    asm volatile("s_wait_asynccnt 0" ::: "memory");
}

// ---------------------------------------------------------------------------
// WMMA FC layer: out[32,N] = relu(X[32,K] @ Wt[N,K]^T + bias[N])
// One wave per 16x16 output tile; block = 4 waves; the block shares a
// double-buffered LDS copy of the full 32-row X k-slab (A operand), loaded
// asynchronously and overlapped with the WMMA on the previous slab.
// B (weights) streams directly from global with prefetch.
// Fragment layouts per CDNA5 ISA 7.12.2 (wave32):
//   A 16x32 f16: lane l -> row M=l%16; VGPR j<4: K=(l/16)*8+2j, j>=4: +16
//   B 32x16 f16: lane l -> col N=l%16; VGPR j<4: K=(l/16)*8+2j, j>=4: +16
//   C/D f32:     VGPR r -> M=(l/16)*8+r, N=l%16
// No divergent control flow (EXEC all-1s for WMMA).
// ---------------------------------------------------------------------------
template<int K, int N, bool RELU>
__global__ void fc_wmma(const float* __restrict__ X,
                        const float* __restrict__ Wt,
                        const float* __restrict__ bias,
                        float* __restrict__ out)
{
    __shared__ float sh[2][32 * LDS_ROW];   // double-buffered A slab (~8.7 KB)

    const int t     = threadIdx.x;                 // 0..127
    const int wave  = blockIdx.x * (blockDim.x >> 5) + (t >> 5);
    const int mtile = wave & 1;                    // 32 rows -> 2 M tiles
    const int ntile = wave >> 1;                   // N/16 tiles
    const int lane  = t & 31;
    const int mn    = lane & 15;
    const int half  = lane >> 4;

    const float* brow = Wt + (size_t)(ntile * 16 + mn) * K;  // B col (Wt row n)

    unsigned int lb0 = (unsigned int)(size_t)(&sh[0][0]);
    unsigned int lb1 = (unsigned int)(size_t)(&sh[1][0]);

    // Prologue: stage slab 0.
    async_stage_slab<K>(X, 0, lb0, t);
    wait_async_all();
    __syncthreads();

    v8f acc = {};
    int cur = 0;
    for (int kk = 0; kk < K; kk += 32) {
        // Kick off the next slab while we compute on the current one.
        if (kk + 32 < K)
            async_stage_slab<K>(X, kk + 32, cur ? lb0 : lb1, t);

        __builtin_prefetch(brow + kk + 256, 0, 1);   // global_prefetch_b8

        // A fragment from LDS (ds_load_b64, conflict-free via padded rows).
        const float* arow = &sh[cur][(mtile * 16 + mn) * LDS_ROW];
        v16h a, b;
        #pragma unroll
        for (int j = 0; j < 4; ++j) {
            float2 p = *(const float2*)(arow + half * 8 + 2 * j);
            a[2 * j]     = (_Float16)p.x;
            a[2 * j + 1] = (_Float16)p.y;
            float2 q = *(const float2*)(arow + 16 + half * 8 + 2 * j);
            a[8 + 2 * j]     = (_Float16)q.x;
            a[8 + 2 * j + 1] = (_Float16)q.y;
        }
        // B fragment direct from global.
        #pragma unroll
        for (int j = 0; j < 8; ++j) {
            float2 p = *(const float2*)(brow + kk + (j >> 2) * 16 + half * 8 + 2 * (j & 3));
            b[2 * j]     = (_Float16)p.x;
            b[2 * j + 1] = (_Float16)p.y;
        }
        acc = __builtin_amdgcn_wmma_f32_16x16x32_f16(
            /*neg_a=*/false, a, /*neg_b=*/false, b,
            /*c_mod=*/(short)0, acc, /*reuse_a=*/false, /*reuse_b=*/false);

        // Drain this wave's async writes, then block-wide handoff.
        wait_async_all();
        __syncthreads();
        cur ^= 1;
    }

    const int col = ntile * 16 + mn;
    const float bv = bias[col];
    #pragma unroll
    for (int r = 0; r < 8; ++r) {
        int row = mtile * 16 + half * 8 + r;
        float v = acc[r] + bv;
        out[(size_t)row * N + col] = RELU ? fmaxf(v, 0.0f) : v;
    }
}

// ---------------------------------------------------------------------------
// fc3 (32x32 -> 32x4) + theta construction. One thread per batch element.
// ---------------------------------------------------------------------------
__global__ void fc3_theta(const float* __restrict__ h2,
                          const float* __restrict__ fw3,
                          const float* __restrict__ fb3,
                          float* __restrict__ theta_ws,
                          float* __restrict__ theta_out)
{
    int b = threadIdx.x;
    if (b >= BATCH) return;
    float s[4];
    #pragma unroll
    for (int j = 0; j < 4; ++j) {
        float acc = fb3[j];
        #pragma unroll
        for (int k = 0; k < 32; ++k)
            acc = fmaf(h2[b * 32 + k], fw3[j * 32 + k], acc);
        s[j] = acc;
    }
    float t00 = 1.0f / (1.0f + expf(-s[0]));
    float t11 = 1.0f / (1.0f + expf(-s[1]));
    float t02 = tanhf(s[2]);
    float t12 = tanhf(s[3]);
    float th[6] = {t00, 0.0f, t02, 0.0f, t11, t12};
    #pragma unroll
    for (int i = 0; i < 6; ++i) {
        theta_ws[b * 6 + i]  = th[i];
        theta_out[b * 6 + i] = th[i];
    }
}

// ---------------------------------------------------------------------------
// affine_grid (align_corners=False) + grid_sample bilinear, zeros padding.
// ---------------------------------------------------------------------------
__global__ void affine_grid_sample(const float* __restrict__ im,
                                   const float* __restrict__ theta,
                                   float* __restrict__ out,
                                   int H, int W)
{
    int idx = blockIdx.x * blockDim.x + threadIdx.x;
    const int total = BATCH * H * W;
    if (idx >= total) return;
    int x = idx % W;
    int t = idx / W;
    int y = t % H;
    int b = t / H;

    const float* th = theta + b * 6;
    float xs = (2.0f * x + 1.0f) / W - 1.0f;
    float ys = (2.0f * y + 1.0f) / H - 1.0f;
    float gx = th[0] * xs + th[1] * ys + th[2];
    float gy = th[3] * xs + th[4] * ys + th[5];

    float ix = ((gx + 1.0f) * W - 1.0f) * 0.5f;
    float iy = ((gy + 1.0f) * H - 1.0f) * 0.5f;
    float x0f = floorf(ix), y0f = floorf(iy);
    int   x0 = (int)x0f,    y0 = (int)y0f;
    float wx1 = ix - x0f, wx0 = 1.0f - wx1;
    float wy1 = iy - y0f, wy0 = 1.0f - wy1;

    const float* src = im + (size_t)b * H * W;
    auto g = [&](int xx, int yy) -> float {
        if (xx < 0 || xx >= W || yy < 0 || yy >= H) return 0.0f;
        return src[yy * W + xx];
    };
    out[idx] = g(x0,     y0    ) * wx0 * wy0
             + g(x0 + 1, y0    ) * wx1 * wy0
             + g(x0,     y0 + 1) * wx0 * wy1
             + g(x0 + 1, y0 + 1) * wx1 * wy1;
}

// ---------------------------------------------------------------------------
extern "C" void kernel_launch(void* const* d_in, const int* in_sizes, int n_in,
                              void* d_out, int out_size, void* d_ws, size_t ws_size,
                              hipStream_t stream)
{
    const float* im  = (const float*)d_in[0];
    const float* w1  = (const float*)d_in[1];
    const float* b1  = (const float*)d_in[2];
    const float* w2  = (const float*)d_in[3];
    const float* b2  = (const float*)d_in[4];
    const float* w3  = (const float*)d_in[5];
    const float* b3  = (const float*)d_in[6];
    const float* w4  = (const float*)d_in[7];
    const float* b4  = (const float*)d_in[8];
    const float* w5  = (const float*)d_in[9];
    const float* b5  = (const float*)d_in[10];
    const float* w6  = (const float*)d_in[11];
    const float* b6  = (const float*)d_in[12];
    const float* fw1 = (const float*)d_in[13];
    const float* fb1 = (const float*)d_in[14];
    const float* fw2 = (const float*)d_in[15];
    const float* fb2 = (const float*)d_in[16];
    const float* fw3 = (const float*)d_in[17];
    const float* fb3 = (const float*)d_in[18];

    float* out       = (float*)d_out;                       // [32*256*256]
    float* theta_out = out + (size_t)BATCH * IMG_H * IMG_W; // + [32*6]

    // Workspace layout (floats), total ~35.4M floats (~142 MB)
    float* ws    = (float*)d_ws;
    float* x1    = ws;                         // 32* 8*256*256 = 16777216
    float* x2    = x1 + 16777216;              // 32* 8*128*128 =  4194304
    float* x3    = x2 + 4194304;               // 32*16*128*128 =  8388608
    float* x4    = x3 + 8388608;               // 32*16* 64* 64 =  2097152
    float* x5    = x4 + 2097152;               // 32*24* 64* 64 =  3145728
    float* xs    = x5 + 3145728;               // 32*24* 32* 32 =   786432
    float* h1    = xs + 786432;                // 32*1024       =    32768
    float* h2    = h1 + 32768;                 // 32*32         =     1024
    float* thws  = h2 + 1024;                  // 32*6          =      192

    const int TB = 256;

    conv3x3_bn_relu<1, 0, 8, 1><<<(32*8*256*256)/TB, TB, 0, stream>>>(im, nullptr, w1, b1, x1, 256, 256);
    conv3x3_bn_relu<8, 1, 8, 2><<<(32*8*128*128)/TB, TB, 0, stream>>>(x1, im,      w2, b2, x2, 256, 256);
    conv3x3_bn_relu<8, 0,16, 1><<<(32*16*128*128)/TB, TB, 0, stream>>>(x2, nullptr, w3, b3, x3, 128, 128);
    conv3x3_bn_relu<16,8,16, 2><<<(32*16*64*64)/TB,   TB, 0, stream>>>(x3, x2,      w4, b4, x4, 128, 128);
    conv3x3_bn_relu<16,0,24, 1><<<(32*24*64*64)/TB,   TB, 0, stream>>>(x4, nullptr, w5, b5, x5, 64, 64);
    conv3x3_bn_relu<24,16,24,2><<<(32*24*32*32)/TB,   TB, 0, stream>>>(x5, x4,      w6, b6, xs, 64, 64);

    // FC head on the WMMA pipe with async-LDS double buffering.
    fc_wmma<24576, 1024, true ><<<32, 128, 0, stream>>>(xs, fw1, fb1, h1);
    fc_wmma<1024,  32,   true ><<<1,  128, 0, stream>>>(h1, fw2, fb2, h2);

    fc3_theta<<<1, 32, 0, stream>>>(h2, fw3, fb3, thws, theta_out);

    affine_grid_sample<<<(32*256*256)/TB, TB, 0, stream>>>(im, thws, out, 256, 256);
}
